// Transformer_67817533604370
// MI455X (gfx1250) — matrix operand order repaired and checked by reading, not compiled, not run on hardware
//
#include <hip/hip_runtime.h>
#include <hip/hip_bf16.h>

// ---------------- model dims ----------------
#define DIMC    512
#define TLEN    2048
#define BATCH   2
#define HEADS   8
#define DHEAD   64
#define MLPC    2048
#define DEPTHC  4
#define MTOT    (BATCH * TLEN)      // 4096 rows
#define QKVN    (3 * DIMC)          // 1536

typedef __attribute__((ext_vector_type(16))) __bf16 v16bf;
typedef __attribute__((ext_vector_type(8)))  float  v8f;

union FragU { unsigned int u[8]; v16bf v; };

__device__ __forceinline__ unsigned short f2bf(float f) {
  unsigned int u = __builtin_bit_cast(unsigned int, f);
  unsigned int r = u + 0x7FFFu + ((u >> 16) & 1u);   // round-to-nearest-even
  return (unsigned short)(r >> 16);
}

// ---------------- LayerNorm: one wave (32 lanes) per 512-elem row ----------------
__global__ __launch_bounds__(256) void ln_kernel(
    const float* __restrict__ x, const float* __restrict__ w,
    const float* __restrict__ b, void* __restrict__ out, int out_is_f32) {
  int wave = threadIdx.x >> 5;
  int lane = threadIdx.x & 31;
  int row  = blockIdx.x * (blockDim.x >> 5) + wave;
  if (row >= MTOT) return;
  const float* xr = x + (long)row * DIMC;
  float v[16];
  float s = 0.f;
  #pragma unroll
  for (int i = 0; i < 16; ++i) { v[i] = xr[lane + 32 * i]; s += v[i]; }
  #pragma unroll
  for (int m = 1; m < 32; m <<= 1) s += __shfl_xor(s, m, 32);
  float mu = s * (1.0f / DIMC);
  float q = 0.f;
  #pragma unroll
  for (int i = 0; i < 16; ++i) { float d = v[i] - mu; q += d * d; }
  #pragma unroll
  for (int m = 1; m < 32; m <<= 1) q += __shfl_xor(q, m, 32);
  float rstd = rsqrtf(q * (1.0f / DIMC) + 1e-5f);
  if (out_is_f32) {
    float* o = (float*)out + (long)row * DIMC;
    #pragma unroll
    for (int i = 0; i < 16; ++i) { int c = lane + 32 * i; o[c] = (v[i] - mu) * rstd * w[c] + b[c]; }
  } else {
    unsigned short* o = (unsigned short*)out + (long)row * DIMC;
    #pragma unroll
    for (int i = 0; i < 16; ++i) { int c = lane + 32 * i; o[c] = f2bf((v[i] - mu) * rstd * w[c] + b[c]); }
  }
}

// ---------------- weight convert + transpose: W[K][N] f32 -> Wt[N][K] bf16 ----------------
__global__ __launch_bounds__(256) void wconv_kernel(
    const float* __restrict__ w, unsigned short* __restrict__ wt, int K, int N) {
  int idx = blockIdx.x * blockDim.x + threadIdx.x;
  if (idx >= K * N) return;
  int n = idx / K, k = idx % K;
  wt[idx] = f2bf(w[(long)k * N + n]);
}

// ---------------- V transpose: qkv bf16 [M][1536] -> vt [B*H][64][T] bf16 ----------------
__global__ __launch_bounds__(256) void vtrans_kernel(
    const unsigned short* __restrict__ qkv, unsigned short* __restrict__ vt) {
  int idx = blockIdx.x * blockDim.x + threadIdx.x;
  if (idx >= BATCH * HEADS * DHEAD * TLEN) return;
  int t = idx % TLEN; int rest = idx / TLEN;
  int d = rest % DHEAD; int bh = rest / DHEAD;
  int b = bh / HEADS, h = bh % HEADS;
  vt[idx] = qkv[((long)(b * TLEN + t)) * QKVN + 2 * DIMC + h * DHEAD + d];
}

// ---------------- bf16 WMMA GEMM: C[M][N] = A[M][K] * Wt[N][K]^T (+bias)(gelu)(resid) ----
// Block tile 64(M) x 128(N); 4 waves; each wave computes 32x64 (2x4 WMMA tiles).
// Global->LDS staging uses CDNA5 async-to-LDS path (ASYNCcnt).
#define GEMM_GELU     1
#define GEMM_RESIDF32 2

__global__ __launch_bounds__(128) void gemm_bf16_kernel(
    const unsigned short* __restrict__ A,   // [M][K] bf16 row-major
    const unsigned short* __restrict__ Wt,  // [N][K] bf16 (transposed weight)
    const float* __restrict__ bias,         // [N] or nullptr
    void* __restrict__ out,                 // bf16 [M][N] or f32 in/out (residual)
    int M, int N, int K, int flags) {
  __shared__ unsigned int As[64 * 16];    // 64 rows x 32 bf16
  __shared__ unsigned int Bs[128 * 16];   // 128 rows x 32 bf16
  int tid  = threadIdx.x;
  int lane = tid & 31, wave = tid >> 5;
  int gm0 = blockIdx.y * 64, gn0 = blockIdx.x * 128;
  int wm = (wave >> 1) * 32, wn = (wave & 1) * 64;
  int lm = lane & 15, hl = lane >> 4;
  v8f acc[2][4] = {};

  // per-thread async staging addresses (advance by 64B per K-chunk)
  const unsigned short* ga[2]; unsigned la[2];
  const unsigned short* gb[4]; unsigned lb[4];
  {
    uint4* As4 = (uint4*)As; uint4* Bs4 = (uint4*)Bs;
    #pragma unroll
    for (int p = 0; p < 2; ++p) {
      int i = tid + p * 128;
      int r = i >> 2, c = i & 3;               // 4 x b128 per 32-elem row
      ga[p] = A + ((long)(gm0 + r)) * K + c * 8;
      la[p] = (unsigned)(unsigned long long)(As4 + i);   // LDS offset in addr[31:0]
    }
    #pragma unroll
    for (int p = 0; p < 4; ++p) {
      int i = tid + p * 128;
      int r = i >> 2, c = i & 3;
      gb[p] = Wt + ((long)(gn0 + r)) * K + c * 8;
      lb[p] = (unsigned)(unsigned long long)(Bs4 + i);
    }
  }

  for (int kc = 0; kc < K; kc += 32) {
    #pragma unroll
    for (int p = 0; p < 2; ++p)
      asm volatile("global_load_async_to_lds_b128 %0, %1, off"
                   :: "v"(la[p]), "v"(ga[p] + kc) : "memory");
    #pragma unroll
    for (int p = 0; p < 4; ++p)
      asm volatile("global_load_async_to_lds_b128 %0, %1, off"
                   :: "v"(lb[p]), "v"(gb[p] + kc) : "memory");
    asm volatile("s_wait_asynccnt 0x0" ::: "memory");
    __syncthreads();

    FragU afr[2], bfr[4];
    #pragma unroll
    for (int mt = 0; mt < 2; ++mt) {
      int m = wm + mt * 16 + lm;
      #pragma unroll
      for (int v = 0; v < 8; ++v) {
        int kk = hl * 16 + 2 * v;                           // contiguous k order: k = 16*hl + s
        afr[mt].u[v] = As[m * 16 + (kk >> 1)];
      }
    }
    #pragma unroll
    for (int nt = 0; nt < 4; ++nt) {
      int n = wn + nt * 16 + lm;
      #pragma unroll
      for (int v = 0; v < 8; ++v) {
        int kk = hl * 16 + 2 * v;                           // same contiguous k order as A
        bfr[nt].u[v] = Bs[n * 16 + (kk >> 1)];
      }
    }
    #pragma unroll
    for (int mt = 0; mt < 2; ++mt)
      #pragma unroll
      for (int nt = 0; nt < 4; ++nt)
        acc[mt][nt] = __builtin_amdgcn_wmma_f32_16x16x32_bf16(
            false, afr[mt].v, false, bfr[nt].v, (short)0, acc[mt][nt], false, false);
    __syncthreads();
  }

  #pragma unroll
  for (int mt = 0; mt < 2; ++mt) {
    #pragma unroll
    for (int nt = 0; nt < 4; ++nt) {
      int col = gn0 + wn + nt * 16 + lm;
      float bv = bias ? bias[col] : 0.f;
      #pragma unroll
      for (int r = 0; r < 8; ++r) {
        int row = gm0 + wm + mt * 16 + r + 8 * hl;          // WMMA C layout
        float v = acc[mt][nt][r] + bv;
        if (flags & GEMM_GELU) v = 0.5f * v * (1.0f + erff(v * 0.70710678118f));
        if (flags & GEMM_RESIDF32) {
          float* o = (float*)out;
          o[(long)row * N + col] += v;
        } else {
          ((unsigned short*)out)[(long)row * N + col] = f2bf(v);
        }
      }
    }
  }
}

// ---------------- block-causal flash attention: one wave per 16-query tile ----------------
__global__ __launch_bounds__(128) void attn_kernel(
    const unsigned short* __restrict__ qkv,  // [M][1536] bf16 (Q|K|V)
    const unsigned short* __restrict__ vt,   // [B*H][64][T] bf16
    unsigned short* __restrict__ o) {        // [M][512] bf16
  __shared__ unsigned short Plds[4][16 * 32];
  int lane = threadIdx.x & 31, wave = threadIdx.x >> 5;
  int g  = blockIdx.x * 4 + wave;
  int qt = g & 127;            // T/16 tiles
  int bh = g >> 7;
  int b = bh >> 3, h = bh & 7;
  int t0 = qt * 16;
  int jmax = (((t0 >> 8) + 1) << 8);   // keys in frames <= query frame
  int lm = lane & 15, hl = lane >> 4;
  unsigned short* P = Plds[wave];

  FragU qa[2];
  {
    long rowbase = ((long)(b * TLEN + t0 + lm)) * QKVN + h * DHEAD;
    #pragma unroll
    for (int c = 0; c < 2; ++c)
      #pragma unroll
      for (int v = 0; v < 8; ++v) {
        int d = c * 32 + hl * 16 + 2 * v;
        qa[c].u[v] = *(const unsigned int*)(qkv + rowbase + d);
      }
  }
  float mrun[8], lrun[8];
  #pragma unroll
  for (int r = 0; r < 8; ++r) { mrun[r] = -1e30f; lrun[r] = 0.f; }
  v8f oacc[4] = {};
  const float scale = 0.125f;  // 64^-0.5

  for (int j = 0; j < jmax; j += 32) {
    v8f s[2] = {};
    #pragma unroll
    for (int half = 0; half < 2; ++half) {
      int jb = j + half * 16;
      long colbase = ((long)(b * TLEN + jb + lm)) * QKVN + DIMC + h * DHEAD;
      #pragma unroll
      for (int c = 0; c < 2; ++c) {
        FragU kb;
        #pragma unroll
        for (int v = 0; v < 8; ++v) {
          int d = c * 32 + hl * 16 + 2 * v;
          kb.u[v] = *(const unsigned int*)(qkv + colbase + d);
        }
        s[half] = __builtin_amdgcn_wmma_f32_16x16x32_bf16(
            false, qa[c].v, false, kb.v, (short)0, s[half], false, false);
      }
    }
    float corr[8];
    #pragma unroll
    for (int r = 0; r < 8; ++r) {
      float a0 = s[0][r] * scale, a1 = s[1][r] * scale;
      float mx = fmaxf(a0, a1);
      #pragma unroll
      for (int msk = 1; msk < 16; msk <<= 1) mx = fmaxf(mx, __shfl_xor(mx, msk, 32));
      float mnew = fmaxf(mrun[r], mx);
      corr[r] = __expf(mrun[r] - mnew);
      float p0 = __expf(a0 - mnew);
      float p1 = __expf(a1 - mnew);
      float sum = p0 + p1;
      #pragma unroll
      for (int msk = 1; msk < 16; msk <<= 1) sum += __shfl_xor(sum, msk, 32);
      lrun[r] = lrun[r] * corr[r] + sum;
      mrun[r] = mnew;
      int row = r + 8 * hl;
      P[row * 32 + lm]      = f2bf(p0);
      P[row * 32 + 16 + lm] = f2bf(p1);
      #pragma unroll
      for (int nt = 0; nt < 4; ++nt) oacc[nt][r] *= corr[r];
    }
    asm volatile("s_wait_dscnt 0" ::: "memory");
    FragU pa;
    #pragma unroll
    for (int v = 0; v < 8; ++v) {
      int kk = hl * 16 + 2 * v;
      pa.u[v] = *(const unsigned int*)(P + lm * 32 + kk);
    }
    #pragma unroll
    for (int nt = 0; nt < 4; ++nt) {
      FragU vb;
      long base = ((long)(bh * DHEAD + nt * 16 + lm)) * TLEN + j;
      #pragma unroll
      for (int v = 0; v < 8; ++v)
        vb.u[v] = *(const unsigned int*)(vt + base + hl * 16 + 2 * v);
      oacc[nt] = __builtin_amdgcn_wmma_f32_16x16x32_bf16(
          false, pa.v, false, vb.v, (short)0, oacc[nt], false, false);
    }
    asm volatile("s_wait_dscnt 0" ::: "memory");
  }
  #pragma unroll
  for (int r = 0; r < 8; ++r) {
    float inv = 1.0f / lrun[r];
    int row = t0 + r + 8 * hl;
    long ob = ((long)(b * TLEN + row)) * DIMC + h * DHEAD;
    #pragma unroll
    for (int nt = 0; nt < 4; ++nt)
      o[ob + nt * 16 + lm] = f2bf(oacc[nt][r] * inv);
  }
}

// ---------------- host orchestration ----------------
extern "C" void kernel_launch(void* const* d_in, const int* in_sizes, int n_in,
                              void* d_out, int out_size, void* d_ws, size_t ws_size,
                              hipStream_t stream) {
  (void)in_sizes; (void)n_in; (void)out_size; (void)ws_size;
  const float* x_in  = (const float*)d_in[0];
  const float* ln1_w = (const float*)d_in[1];
  const float* ln1_b = (const float*)d_in[2];
  const float* qkv_w = (const float*)d_in[3];
  const float* out_w = (const float*)d_in[4];
  const float* out_b = (const float*)d_in[5];
  const float* ln2_w = (const float*)d_in[6];
  const float* ln2_b = (const float*)d_in[7];
  const float* ff_w1 = (const float*)d_in[8];
  const float* ff_b1 = (const float*)d_in[9];
  const float* ff_w2 = (const float*)d_in[10];
  const float* ff_b2 = (const float*)d_in[11];
  const float* lnf_w = (const float*)d_in[12];
  const float* lnf_b = (const float*)d_in[13];

  char* ws = (char*)d_ws;
  size_t off = 0;
  auto carve = [&](size_t bytes) -> void* {
    void* p = ws + off;
    off += (bytes + 255) & ~(size_t)255;
    return p;
  };
  float*          x    = (float*)carve((size_t)MTOT * DIMC * 4);
  unsigned short* xn   = (unsigned short*)carve((size_t)MTOT * DIMC * 2);
  unsigned short* qkvb = (unsigned short*)carve((size_t)MTOT * QKVN * 2);
  unsigned short* vtb  = (unsigned short*)carve((size_t)BATCH * HEADS * DHEAD * TLEN * 2);
  unsigned short* ob   = (unsigned short*)carve((size_t)MTOT * DIMC * 2);
  unsigned short* hb   = (unsigned short*)carve((size_t)MTOT * MLPC * 2);
  unsigned short* wqkvT = (unsigned short*)carve((size_t)DEPTHC * QKVN * DIMC * 2);
  unsigned short* woutT = (unsigned short*)carve((size_t)DEPTHC * DIMC * DIMC * 2);
  unsigned short* wff1T = (unsigned short*)carve((size_t)DEPTHC * MLPC * DIMC * 2);
  unsigned short* wff2T = (unsigned short*)carve((size_t)DEPTHC * DIMC * MLPC * 2);

  hipMemcpyAsync(x, x_in, (size_t)MTOT * DIMC * 4, hipMemcpyDeviceToDevice, stream);

  // convert all weights (f32 [K][N]) -> transposed bf16 [N][K]
  for (int i = 0; i < DEPTHC; ++i) {
    wconv_kernel<<<(DIMC * QKVN) / 256, 256, 0, stream>>>(
        qkv_w + (long)i * DIMC * QKVN, wqkvT + (long)i * QKVN * DIMC, DIMC, QKVN);
    wconv_kernel<<<(DIMC * DIMC) / 256, 256, 0, stream>>>(
        out_w + (long)i * DIMC * DIMC, woutT + (long)i * DIMC * DIMC, DIMC, DIMC);
    wconv_kernel<<<(DIMC * MLPC) / 256, 256, 0, stream>>>(
        ff_w1 + (long)i * DIMC * MLPC, wff1T + (long)i * MLPC * DIMC, DIMC, MLPC);
    wconv_kernel<<<(MLPC * DIMC) / 256, 256, 0, stream>>>(
        ff_w2 + (long)i * MLPC * DIMC, wff2T + (long)i * DIMC * MLPC, MLPC, DIMC);
  }

  for (int i = 0; i < DEPTHC; ++i) {
    // attention block
    ln_kernel<<<MTOT / 8, 256, 0, stream>>>(x, ln1_w + i * DIMC, ln1_b + i * DIMC, xn, 0);
    gemm_bf16_kernel<<<dim3(QKVN / 128, MTOT / 64), 128, 0, stream>>>(
        xn, wqkvT + (long)i * QKVN * DIMC, nullptr, qkvb, MTOT, QKVN, DIMC, 0);
    vtrans_kernel<<<(BATCH * HEADS * DHEAD * TLEN) / 256, 256, 0, stream>>>(qkvb, vtb);
    attn_kernel<<<(BATCH * HEADS * (TLEN / 16)) / 4, 128, 0, stream>>>(qkvb, vtb, ob);
    gemm_bf16_kernel<<<dim3(DIMC / 128, MTOT / 64), 128, 0, stream>>>(
        ob, woutT + (long)i * DIMC * DIMC, out_b + i * DIMC, x, MTOT, DIMC, DIMC, GEMM_RESIDF32);
    // feedforward block
    ln_kernel<<<MTOT / 8, 256, 0, stream>>>(x, ln2_w + i * DIMC, ln2_b + i * DIMC, xn, 0);
    gemm_bf16_kernel<<<dim3(MLPC / 128, MTOT / 64), 128, 0, stream>>>(
        xn, wff1T + (long)i * MLPC * DIMC, ff_b1 + i * MLPC, hb, MTOT, MLPC, DIMC, GEMM_GELU);
    gemm_bf16_kernel<<<dim3(DIMC / 128, MTOT / 64), 128, 0, stream>>>(
        hb, wff2T + (long)i * DIMC * MLPC, ff_b2 + i * DIMC, x, MTOT, DIMC, MLPC, GEMM_RESIDF32);
  }
  ln_kernel<<<MTOT / 8, 256, 0, stream>>>(x, lnf_w, lnf_b, d_out, 1);
}
